// Decoder_28930899706149
// MI455X (gfx1250) — matrix operand order repaired and checked by reading, not compiled, not run on hardware
//
#include <hip/hip_runtime.h>
#include <hip/hip_bf16.h>
#include <math.h>

// Problem dims (compile-time constants from the reference)
#define B_  64
#define S_  2048
#define H_  1024
#define E_  512
#define V_  32000
#define G4_ 4096        // 4*H
#define XD_ 1536        // E + H  (concat [emb, context])
#define PSTR_ 1028      // per-chunk partial stride: 1024 ctx + m + sum (+pad)

typedef __attribute__((ext_vector_type(16))) __bf16 v16bf;
typedef __attribute__((ext_vector_type(8)))  float  v8f;

// ======================================================================
// WMMA bf16 fragment helpers (CDNA5 V_WMMA_F32_16X16X32_BF16)
//
// A (16x32, 16-bit) per-lane layout used by this code (contiguous k, not the ISA table order):
//   lane l holds row M = l%16; half lh = l/16 selects K-subsets:
//   elems 0..15 -> K = k0 + 16*lh + e       (contiguous 16 floats)
// B (32x16, 16-bit) per-lane layout:
//   lane l holds col N = l%16; kh = l/16:
//   elems 0..15 -> K = k0 + 16*kh + e       (contiguous 16 floats)
// ======================================================================
__device__ __forceinline__ v16bf load_frag_a(const float* __restrict__ A, int lda,
                                             int r0, int k0, int lane) {
    const int r  = lane & 15;
    const int lh = lane >> 4;
    const float* p = A + (size_t)(r0 + r) * lda + k0 + 16 * lh;
    v16bf a;
#pragma unroll
    for (int e = 0; e < 8; ++e) a[e] = (__bf16)p[e];
#pragma unroll
    for (int e = 0; e < 8; ++e) a[8 + e] = (__bf16)p[8 + e];
    return a;
}

__device__ __forceinline__ v16bf load_frag_b(const float* __restrict__ W, int ldb,
                                             int n0, int k0, int lane) {
    const int n  = lane & 15;
    const int kh = lane >> 4;
    const float* p = W + (size_t)(n0 + n) * ldb + k0 + 16 * kh;
    v16bf b;
#pragma unroll
    for (int e = 0; e < 16; ++e) b[e] = (__bf16)p[e];
    return b;
}

// ======================================================================
// 1) Embedding gather: xbuf[b, 0:E] = emb_W[input_batch[b]]
// ======================================================================
__global__ __launch_bounds__(256) void embed_kernel(const int* __restrict__ idx,
                                                    const float* __restrict__ emb_W,
                                                    float* __restrict__ xbuf) {
    const int i = blockIdx.x * 256 + threadIdx.x;    // B*E
    const int b = i >> 9;                            // /512
    const int e = i & 511;
    xbuf[(size_t)b * XD_ + e] = emb_W[(size_t)idx[b] * E_ + e];
}

// ======================================================================
// 2) Flash-style attention pass: single read of encoder_outputs.
//    scores[b,s] = prev_h@Wh + enc[s,b]@We + attn_b; the first & last
//    terms are constant in s, so softmax(scores) == softmax(enc@We):
//    we drop them (shift invariance). Online softmax accumulates the
//    unnormalized context per wave entirely in registers.
//    grid = B*16 chunks, 256 threads; wave w owns s in [base+16w, +16).
// ======================================================================
__global__ __launch_bounds__(256) void attn_partial_kernel(const float* __restrict__ enc,
                                                           const float* __restrict__ attn_W,
                                                           float* __restrict__ e_buf,
                                                           float* __restrict__ part) {
    const int blk   = blockIdx.x;          // b*16 + chunk
    const int b     = blk >> 4;
    const int chunk = blk & 15;
    const int wave  = threadIdx.x >> 5;
    const int lane  = threadIdx.x & 31;

    // We = attn_W[H:2H]; lane holds h = (i*32+lane)*4 + c, i=0..7
    const float4* We4 = (const float4*)(attn_W + H_);
    float4 we[8];
#pragma unroll
    for (int i = 0; i < 8; ++i) we[i] = We4[i * 32 + lane];

    float  m = -INFINITY, ssum = 0.f;
    float4 acc[8];
#pragma unroll
    for (int i = 0; i < 8; ++i) acc[i] = make_float4(0.f, 0.f, 0.f, 0.f);

    const int s_base = chunk * 128 + wave * 16;
    for (int si = 0; si < 16; ++si) {
        const int s = s_base + si;
        const float4* row = (const float4*)(enc + ((size_t)s * B_ + b) * H_);
        float4 v[8];
        float partial = 0.f;
#pragma unroll
        for (int i = 0; i < 8; ++i) {
            v[i] = row[i * 32 + lane];                       // global_load_b128
            partial += v[i].x * we[i].x + v[i].y * we[i].y
                     + v[i].z * we[i].z + v[i].w * we[i].w;
        }
#pragma unroll
        for (int off = 16; off; off >>= 1)                   // butterfly: all lanes get e
            partial += __shfl_xor(partial, off, 32);
        const float e = partial;
        if (lane == 0) e_buf[(size_t)b * S_ + s] = e;

        const float nm = fmaxf(m, e);
        const float sc = __expf(m - nm);                     // first iter: exp(-inf)=0
        const float p  = __expf(e - nm);
        ssum = ssum * sc + p;
#pragma unroll
        for (int i = 0; i < 8; ++i) {
            acc[i].x = acc[i].x * sc + p * v[i].x;
            acc[i].y = acc[i].y * sc + p * v[i].y;
            acc[i].z = acc[i].z * sc + p * v[i].z;
            acc[i].w = acc[i].w * sc + p * v[i].w;
        }
        m = nm;
    }

    // combine 8 waves within the block (LDS float atomics = ds_add_f32)
    __shared__ float lds_m[8], lds_s[8];
    __shared__ float lds_ctx[H_];
    for (int h = threadIdx.x; h < H_; h += 256) lds_ctx[h] = 0.f;
    if (lane == 0) { lds_m[wave] = m; lds_s[wave] = ssum; }
    __syncthreads();
    float M = lds_m[0];
#pragma unroll
    for (int w = 1; w < 8; ++w) M = fmaxf(M, lds_m[w]);
    const float scw = __expf(m - M);
#pragma unroll
    for (int i = 0; i < 8; ++i) {
        const int hb = (i * 32 + lane) * 4;
        atomicAdd(&lds_ctx[hb + 0], acc[i].x * scw);
        atomicAdd(&lds_ctx[hb + 1], acc[i].y * scw);
        atomicAdd(&lds_ctx[hb + 2], acc[i].z * scw);
        atomicAdd(&lds_ctx[hb + 3], acc[i].w * scw);
    }
    __syncthreads();
    float* out = part + (size_t)blk * PSTR_;
    for (int h = threadIdx.x; h < H_; h += 256) out[h] = lds_ctx[h];
    if (threadIdx.x == 0) {
        float bs = 0.f;
#pragma unroll
        for (int w = 0; w < 8; ++w) bs += lds_s[w] * __expf(lds_m[w] - M);
        out[1024] = M;
        out[1025] = bs;
    }
}

// ======================================================================
// 3) Merge 16 chunk partials per batch -> context (into xbuf[:, E:]),
//    and save (row max, denom) for the attn_weights kernel.
// ======================================================================
__global__ __launch_bounds__(256) void attn_combine_kernel(const float* __restrict__ part,
                                                           float* __restrict__ xbuf,
                                                           float* __restrict__ mdenom) {
    const int b = blockIdx.x;
    const int tid = threadIdx.x;
    float pm[16], ps[16];
#pragma unroll
    for (int c = 0; c < 16; ++c) {
        const float* pp = part + (size_t)(b * 16 + c) * PSTR_;
        pm[c] = pp[1024];
        ps[c] = pp[1025];
    }
    float M = pm[0];
#pragma unroll
    for (int c = 1; c < 16; ++c) M = fmaxf(M, pm[c]);
    float denom = 0.f;
    float w[16];
#pragma unroll
    for (int c = 0; c < 16; ++c) { w[c] = __expf(pm[c] - M); denom += w[c] * ps[c]; }
    const float inv = 1.f / denom;
#pragma unroll
    for (int k = 0; k < 4; ++k) {
        const int h = tid + k * 256;
        float ctx = 0.f;
#pragma unroll
        for (int c = 0; c < 16; ++c)
            ctx += w[c] * part[(size_t)(b * 16 + c) * PSTR_ + h];
        xbuf[(size_t)b * XD_ + E_ + h] = ctx * inv;
    }
    if (tid == 0) { mdenom[2 * b] = M; mdenom[2 * b + 1] = denom; }
}

// ======================================================================
// 4) attn_weights[b,s] = exp(e - M_b)/denom_b   (B*S elements)
// ======================================================================
__global__ __launch_bounds__(256) void attn_weights_kernel(const float* __restrict__ e_buf,
                                                           const float* __restrict__ mdenom,
                                                           float* __restrict__ attn_out) {
    const int idx = blockIdx.x * 256 + threadIdx.x;
    const int b = idx >> 11;                       // /2048
    const float M  = mdenom[2 * b];
    const float iv = 1.f / mdenom[2 * b + 1];
    attn_out[idx] = __expf(e_buf[idx] - M) * iv;
}

// ======================================================================
// 5) Gates GEMM (WMMA bf16): gates = [x, prev_h] @ [W_ih, W_hh]^T + biases
//    64x4096, K = 1536 (x/W_ih) + 1024 (prev_h/W_hh). 1 tile per wave.
// ======================================================================
__global__ __launch_bounds__(256) void gates_gemm_kernel(const float* __restrict__ xbuf,
                                                         const float* __restrict__ prev_h,
                                                         const float* __restrict__ W_ih,
                                                         const float* __restrict__ W_hh,
                                                         const float* __restrict__ b_ih,
                                                         const float* __restrict__ b_hh,
                                                         float* __restrict__ gates) {
    const int wave = threadIdx.x >> 5, lane = threadIdx.x & 31;
    const int wg = blockIdx.x * 8 + wave;          // 1024 tiles
    const int m0 = (wg & 3) * 16;                  // batch tile
    const int n0 = (wg >> 2) * 16;                 // gate-col tile
    v8f c;
#pragma unroll
    for (int e = 0; e < 8; ++e) c[e] = 0.f;

    for (int k0 = 0; k0 < XD_; k0 += 32) {
        v16bf a = load_frag_a(xbuf, XD_, m0, k0, lane);
        v16bf b = load_frag_b(W_ih, XD_, n0, k0, lane);
        c = __builtin_amdgcn_wmma_f32_16x16x32_bf16(false, a, false, b, (short)0, c, false, false);
    }
    for (int k0 = 0; k0 < H_; k0 += 32) {
        v16bf a = load_frag_a(prev_h, H_, m0, k0, lane);
        v16bf b = load_frag_b(W_hh, H_, n0, k0, lane);
        c = __builtin_amdgcn_wmma_f32_16x16x32_bf16(false, a, false, b, (short)0, c, false, false);
    }
    const int col  = n0 + (lane & 15);
    const float bias = b_ih[col] + b_hh[col];
#pragma unroll
    for (int r = 0; r < 8; ++r) {
        const int row = m0 + r + 8 * (lane >> 4);   // C/D layout: M = r + 8*(lane/16)
        gates[(size_t)row * G4_ + col] = c[r] + bias;
    }
}

// ======================================================================
// 6) LSTM elementwise: split gates -> i,f,g,o; cell, hidden
// ======================================================================
__global__ __launch_bounds__(256) void lstm_kernel(const float* __restrict__ gates,
                                                   const float* __restrict__ prev_c,
                                                   float* __restrict__ hidden,
                                                   float* __restrict__ cell) {
    const int i = blockIdx.x * 256 + threadIdx.x;  // B*H
    const int b = i >> 10, h = i & 1023;
    const float gi = gates[(size_t)b * G4_ + h];
    const float gf = gates[(size_t)b * G4_ + H_ + h];
    const float gg = gates[(size_t)b * G4_ + 2 * H_ + h];
    const float go = gates[(size_t)b * G4_ + 3 * H_ + h];
    const float si = 1.f / (1.f + __expf(-gi));
    const float sf = 1.f / (1.f + __expf(-gf));
    const float so = 1.f / (1.f + __expf(-go));
    const float tg = tanhf(gg);
    const float cc = sf * prev_c[i] + si * tg;
    hidden[i] = so * tanhf(cc);
    cell[i]   = cc;
}

// ======================================================================
// 7) Logits GEMM (WMMA bf16): logits = hidden @ out_W^T + out_b
//    64x32000, K=1024. 8000 tiles, 1 per wave, 1000 blocks.
// ======================================================================
__global__ __launch_bounds__(256) void logits_gemm_kernel(const float* __restrict__ hidden,
                                                          const float* __restrict__ out_W,
                                                          const float* __restrict__ out_b,
                                                          float* __restrict__ logits) {
    const int wave = threadIdx.x >> 5, lane = threadIdx.x & 31;
    const int wg = blockIdx.x * 8 + wave;          // 8000 tiles
    const int m0 = (wg & 3) * 16;                  // batch tile
    const int n0 = (wg >> 2) * 16;                 // vocab tile
    v8f c;
#pragma unroll
    for (int e = 0; e < 8; ++e) c[e] = 0.f;

    for (int k0 = 0; k0 < H_; k0 += 32) {
        v16bf a = load_frag_a(hidden, H_, m0, k0, lane);
        v16bf b = load_frag_b(out_W, H_, n0, k0, lane);
        c = __builtin_amdgcn_wmma_f32_16x16x32_bf16(false, a, false, b, (short)0, c, false, false);
    }
    const int col = n0 + (lane & 15);
    const float bias = out_b[col];
#pragma unroll
    for (int r = 0; r < 8; ++r) {
        const int row = m0 + r + 8 * (lane >> 4);
        logits[(size_t)row * V_ + col] = c[r] + bias;
    }
}

// ======================================================================
// 8) log_softmax over V per row (max, sumexp, write) — block reductions
// ======================================================================
__global__ __launch_bounds__(256) void logsoftmax_kernel(const float* __restrict__ logits,
                                                         float* __restrict__ out) {
    const int b = blockIdx.x, tid = threadIdx.x;
    const float* row = logits + (size_t)b * V_;
    __shared__ float red[8];

    float m = -INFINITY;
    for (int v = tid; v < V_; v += 256) m = fmaxf(m, row[v]);
#pragma unroll
    for (int off = 16; off; off >>= 1) m = fmaxf(m, __shfl_xor(m, off, 32));
    if ((tid & 31) == 0) red[tid >> 5] = m;
    __syncthreads();
    m = red[0];
#pragma unroll
    for (int w = 1; w < 8; ++w) m = fmaxf(m, red[w]);
    __syncthreads();

    float s = 0.f;
    for (int v = tid; v < V_; v += 256) s += __expf(row[v] - m);
#pragma unroll
    for (int off = 16; off; off >>= 1) s += __shfl_xor(s, off, 32);
    if ((tid & 31) == 0) red[tid >> 5] = s;
    __syncthreads();
    s = 0.f;
#pragma unroll
    for (int w = 0; w < 8; ++w) s += red[w];
    const float lse = m + logf(s);

    float* orow = out + (size_t)b * V_;
    for (int v = tid; v < V_; v += 256) orow[v] = row[v] - lse;
}

// ======================================================================
// Launch
// ======================================================================
extern "C" void kernel_launch(void* const* d_in, const int* in_sizes, int n_in,
                              void* d_out, int out_size, void* d_ws, size_t ws_size,
                              hipStream_t stream) {
    const int*   input_batch = (const int*)  d_in[0];
    const float* prev_h      = (const float*)d_in[1];
    const float* prev_c      = (const float*)d_in[2];
    const float* enc         = (const float*)d_in[3];
    const float* emb_W       = (const float*)d_in[4];
    const float* attn_W      = (const float*)d_in[5];
    // d_in[6] = attn_b : constant shift per softmax row -> cancels, unused.
    const float* W_ih        = (const float*)d_in[7];
    const float* W_hh        = (const float*)d_in[8];
    const float* b_ih        = (const float*)d_in[9];
    const float* b_hh        = (const float*)d_in[10];
    const float* out_W       = (const float*)d_in[11];
    const float* out_b       = (const float*)d_in[12];

    float* out_logp = (float*)d_out;                          // B*V
    float* out_hid  = out_logp + (size_t)B_ * V_;             // B*H
    float* out_cell = out_hid  + (size_t)B_ * H_;             // B*H
    float* out_attn = out_cell + (size_t)B_ * H_;             // B*S

    float* ws      = (float*)d_ws;
    float* e_buf   = ws;                             // B*S            = 131072
    float* part    = e_buf  + (size_t)B_ * S_;       // 1024*PSTR_     = 1052672
    float* xbuf    = part   + (size_t)1024 * PSTR_;  // B*XD_          = 98304
    float* mdenom  = xbuf   + (size_t)B_ * XD_;      // 2*B            = 128
    float* gates   = mdenom + 128;                   // B*4H           = 262144
    float* logits  = gates  + (size_t)B_ * G4_;      // B*V            = 2048000
    (void)in_sizes; (void)n_in; (void)out_size; (void)ws_size; (void)logits;

    embed_kernel       <<<(B_ * E_) / 256, 256, 0, stream>>>(input_batch, emb_W, xbuf);
    attn_partial_kernel<<<B_ * 16,         256, 0, stream>>>(enc, attn_W, e_buf, part);
    attn_combine_kernel<<<B_,              256, 0, stream>>>(part, xbuf, mdenom);
    attn_weights_kernel<<<(B_ * S_) / 256, 256, 0, stream>>>(e_buf, mdenom, out_attn);
    gates_gemm_kernel  <<<128,             256, 0, stream>>>(xbuf, prev_h, W_ih, W_hh, b_ih, b_hh, gates);
    lstm_kernel        <<<(B_ * H_) / 256, 256, 0, stream>>>(gates, prev_c, out_hid, out_cell);
    logits_gemm_kernel <<<1000,            256, 0, stream>>>(out_hid, out_W, out_b, logits);
    logsoftmax_kernel  <<<B_,              256, 0, stream>>>(logits, out_logp);
}
